// LSTMClassifier_81277961110113
// MI455X (gfx1250) — matrix operand order repaired and checked by reading, not compiled, not run on hardware
//
#include <hip/hip_runtime.h>
#include <hip/hip_bf16.h>
#include <type_traits>

typedef __attribute__((ext_vector_type(16))) __bf16 v16bf;
typedef __attribute__((ext_vector_type(8)))  __bf16 v8bf;
typedef __attribute__((ext_vector_type(8)))  float  v8f;

#define BB 256
#define TT 1024
#define DD 64
#define HH 128
#define GG 512   // 4*H
#define CC 10

// ---------------------------------------------------------------------------
// Branchless transcendentals. gfx1250 has V_TANH_F32; prefer the builtin.
// ---------------------------------------------------------------------------
__device__ __forceinline__ float fast_tanh(float x) {
#if __has_builtin(__builtin_amdgcn_tanhf)
    return __builtin_amdgcn_tanhf(x);
#else
    // tanh(x) = 1 - 2/(e^{2x}+1); e^{2x} = 2^{2x*log2(e)}.  inf/0 saturate
    // correctly, so no range branches needed.
    float t = __builtin_amdgcn_exp2f(2.885390082f * x);
    return 1.0f - 2.0f * __builtin_amdgcn_rcpf(t + 1.0f);
#endif
}
__device__ __forceinline__ float fast_sigmoid(float x) {
    return fmaf(0.5f, fast_tanh(0.5f * x), 0.5f);
}

// ---------------------------------------------------------------------------
// B-operand fragment (bf16 32x16, K x N): lanes 0-15 hold K=kk..kk+15 for
// column N=n0+lane, lanes 16-31 hold K=kk+16..kk+31.  With W row-major [N][K]
// (g = a @ W^T) a lane's fragment is 16 contiguous bf16 => one 32-byte load.
// ---------------------------------------------------------------------------
__device__ __forceinline__ v16bf load_B_frag(const __bf16* __restrict__ W, int K,
                                             int n0, int lane, int kk) {
    int n  = n0 + (lane & 15);
    int kb = kk + ((lane >> 4) << 4);
    return *(const v16bf*)(W + (size_t)n * K + kb);
}

// ---------------------------------------------------------------------------
// A-operand fragment (bf16 16x32, M x K): lane m=lane&15; lanes 0-15 carry
// K=kk+0..15; lanes 16-31 carry K=kk+16..31 (matches B fragment order).
// Two contiguous 16-byte chunks from a row-major LDS tile.
// ---------------------------------------------------------------------------
__device__ __forceinline__ v16bf load_A_frag(const __bf16* __restrict__ tile,
                                             int stride, int lane, int kk) {
    int m  = lane & 15;
    int hi = (lane >> 4) * 16;
    const __bf16* base = tile + m * stride + kk;
    v8bf a = *(const v8bf*)(base + hi);
    v8bf b = *(const v8bf*)(base + 8 + hi);
    v16bf r;
#pragma unroll
    for (int i = 0; i < 8; ++i) { r[i] = a[i]; r[8 + i] = b[i]; }
    return r;
}

// ---------------------------------------------------------------------------
// fp32 -> bf16 weight conversion (one-time)
// ---------------------------------------------------------------------------
__global__ void cvt_bf16_kernel(const float* __restrict__ src,
                                __bf16* __restrict__ dst, int n) {
    int i = blockIdx.x * blockDim.x + threadIdx.x;
    if (i < n) dst[i] = (__bf16)src[i];
}

// ---------------------------------------------------------------------------
// Persistent LSTM scan: one block = 16 batch rows for all T steps.
// 32 waves (1024 threads); wave w owns gate columns [16w, 16w+16).
// h (bf16) and c (fp32) live in LDS for the entire sequence.
// Software-pipelined input: x_{t+1} staged to LDS during the elementwise
// phase of step t; global loads for x_{t+2} issued immediately after.
// ---------------------------------------------------------------------------
template <int DIN, bool IN_BF16, bool STORE_ALL>
__launch_bounds__(1024)
__global__ void lstm_scan_kernel(const void* __restrict__ input,
                                 const __bf16* __restrict__ Wih,
                                 const __bf16* __restrict__ Whh,
                                 const float* __restrict__ bih,
                                 const float* __restrict__ bhh,
                                 __bf16* __restrict__ h_all,
                                 float* __restrict__ h_last) {
    constexpr int XST  = DIN + 8;       // padded strides (bf16): rows stay 16B-aligned
    constexpr int HSTR = HH + 8;
    constexpr int GST  = GG + 8;        // fp32 gate buffer stride (bank pad)
    constexpr int EPT  = (16 * DIN) / 1024;   // input elems per thread (1 or 2)
    constexpr int KIN  = DIN / 32;            // k-steps for input GEMM

    using InT = typename std::conditional<IN_BF16, __bf16, float>::type;

    __shared__ __align__(16) __bf16 xtile[16 * XST];
    __shared__ __align__(16) __bf16 htile[16 * HSTR];
    __shared__ float gbuf[16 * GST];
    __shared__ float ctile[16 * HH];

    const int tid  = threadIdx.x;
    const int lane = tid & 31;
    const int wave = tid >> 5;
    const int n0   = wave * 16;          // gate-column tile owned by this wave
    const int b0   = blockIdx.x * 16;    // batch-row tile owned by this block

    // ---- hoist all weight B-fragments into registers for the whole T loop
    v16bf wihf[KIN];
#pragma unroll
    for (int s = 0; s < KIN; ++s) wihf[s] = load_B_frag(Wih, DIN, n0, lane, s * 32);
    v16bf whhf[4];
#pragma unroll
    for (int s = 0; s < 4; ++s) whhf[s] = load_B_frag(Whh, HH, n0, lane, s * 32);

    // ---- bias for this lane's gate column, folded into the WMMA C-operand
    const int ncol = n0 + (lane & 15);
    const float bw = bih[ncol] + bhh[ncol];

    // ---- per-thread input pipeline state
    const InT* in = (const InT*)input;
    const InT* inbase[EPT];
    int        ldsoff[EPT];
    InT        xreg[EPT];
#pragma unroll
    for (int i = 0; i < EPT; ++i) {
        int e = tid + i * 1024;
        int r = e / DIN, c = e % DIN;
        inbase[i] = in + (size_t)(b0 + r) * TT * DIN + c;
        ldsoff[i] = r * XST + c;
        xreg[i]   = inbase[i][0];                    // x_0
    }

    // ---- per-thread elementwise coordinates (2 cells per thread)
    int    er[2], ej[2];
    size_t hoff[2];
#pragma unroll
    for (int i = 0; i < 2; ++i) {
        int e  = tid + i * 1024;
        er[i]  = e >> 7;
        ej[i]  = e & (HH - 1);
        hoff[i] = (size_t)(b0 + er[i]) * TT * HH + ej[i];
    }

    // ---- init state, stage x_0, issue loads for x_1
    for (int e = tid; e < 16 * HH; e += 1024)   ctile[e] = 0.0f;
    for (int e = tid; e < 16 * HSTR; e += 1024) htile[e] = (__bf16)0.0f;
#pragma unroll
    for (int i = 0; i < EPT; ++i) xtile[ldsoff[i]] = (__bf16)xreg[i];
#pragma unroll
    for (int i = 0; i < EPT; ++i) xreg[i] = inbase[i][DIN];   // x_1
    __syncthreads();

    for (int t = 0; t < TT; ++t) {
        // ---- g = bias + x_t @ Wih^T + h @ Whh^T  (bf16 WMMA, fp32 accum)
        v8f acc;
#pragma unroll
        for (int vr = 0; vr < 8; ++vr) acc[vr] = bw;
#pragma unroll
        for (int s = 0; s < KIN; ++s) {
            v16bf a = load_A_frag(xtile, XST, lane, s * 32);
            acc = __builtin_amdgcn_wmma_f32_16x16x32_bf16(
                false, a, false, wihf[s], (short)0, acc, false, false);
        }
#pragma unroll
        for (int s = 0; s < 4; ++s) {
            v16bf a = load_A_frag(htile, HSTR, lane, s * 32);
            acc = __builtin_amdgcn_wmma_f32_16x16x32_bf16(
                false, a, false, whhf[s], (short)0, acc, false, false);
        }

        // ---- scatter D fragment to gate buffer (vr->row, lane%16->col)
        {
            int mb = (lane >> 4) * 8;
#pragma unroll
            for (int vr = 0; vr < 8; ++vr) gbuf[(mb + vr) * GST + ncol] = acc[vr];
        }
        __syncthreads();

        // ---- fused gate nonlinearity + state update (fp32, branchless)
#pragma unroll
        for (int i = 0; i < 2; ++i) {
            int r = er[i], j = ej[i];
            const float* grow = &gbuf[r * GST];
            float iv = fast_sigmoid(grow[j]);
            float fv = fast_sigmoid(grow[HH + j]);
            float gv = fast_tanh(grow[2 * HH + j]);
            float ov = fast_sigmoid(grow[3 * HH + j]);
            int   ce = tid + i * 1024;
            float c  = fmaf(fv, ctile[ce], iv * gv);
            float h  = ov * fast_tanh(c);
            ctile[ce] = c;
            htile[r * HSTR + j] = (__bf16)h;
            if (STORE_ALL) {
                h_all[hoff[i] + (size_t)t * HH] = (__bf16)h;
            } else if (t == TT - 1) {
                h_last[(size_t)(b0 + r) * HH + j] = h;
            }
        }

        // ---- stage x_{t+1} (held in regs), issue loads for x_{t+2}
        if (t + 1 < TT) {
#pragma unroll
            for (int i = 0; i < EPT; ++i) xtile[ldsoff[i]] = (__bf16)xreg[i];
            if (t + 2 < TT) {
#pragma unroll
                for (int i = 0; i < EPT; ++i)
                    xreg[i] = inbase[i][(size_t)(t + 2) * DIN];
                if (t + 3 < TT)
                    __builtin_prefetch(inbase[0] + (size_t)(t + 3) * DIN, 0, 1);
            }
        }
        __syncthreads();
    }
}

// ---------------------------------------------------------------------------
// Classifier head: out = h2_last @ W_head^T + b_head   (256x128 x 128x10)
// ---------------------------------------------------------------------------
__global__ void head_kernel(const float* __restrict__ h2,
                            const float* __restrict__ Wh,
                            const float* __restrict__ bh,
                            float* __restrict__ out) {
    int b = blockIdx.x * blockDim.x + threadIdx.x;
    if (b >= BB) return;
#pragma unroll
    for (int cls = 0; cls < CC; ++cls) {
        float s = bh[cls];
#pragma unroll 8
        for (int j = 0; j < HH; ++j) s += h2[b * HH + j] * Wh[cls * HH + j];
        out[b * CC + cls] = s;
    }
}

// ---------------------------------------------------------------------------
extern "C" void kernel_launch(void* const* d_in, const int* in_sizes, int n_in,
                              void* d_out, int out_size, void* d_ws, size_t ws_size,
                              hipStream_t stream) {
    const float* x     = (const float*)d_in[0];
    const float* Wih0  = (const float*)d_in[1];
    const float* Whh0  = (const float*)d_in[2];
    const float* bih0  = (const float*)d_in[3];
    const float* bhh0  = (const float*)d_in[4];
    const float* Wih1  = (const float*)d_in[5];
    const float* Whh1  = (const float*)d_in[6];
    const float* bih1  = (const float*)d_in[7];
    const float* bhh1  = (const float*)d_in[8];
    const float* Whead = (const float*)d_in[9];
    const float* bhead = (const float*)d_in[10];

    // workspace layout (bytes)
    char* ws = (char*)d_ws;
    __bf16* wih0b = (__bf16*)(ws + 0);        // 512*64*2  = 65536
    __bf16* whh0b = (__bf16*)(ws + 65536);    // 512*128*2 = 131072
    __bf16* wih1b = (__bf16*)(ws + 196608);   // 131072
    __bf16* whh1b = (__bf16*)(ws + 327680);   // 131072
    float*  h2l   = (float*)(ws + 458752);    // 256*128*4 = 131072
    __bf16* h1    = (__bf16*)(ws + 589824);   // 256*1024*128*2 = 64 MiB

    cvt_bf16_kernel<<<(GG * DD + 255) / 256, 256, 0, stream>>>(Wih0, wih0b, GG * DD);
    cvt_bf16_kernel<<<(GG * HH + 255) / 256, 256, 0, stream>>>(Whh0, whh0b, GG * HH);
    cvt_bf16_kernel<<<(GG * HH + 255) / 256, 256, 0, stream>>>(Wih1, wih1b, GG * HH);
    cvt_bf16_kernel<<<(GG * HH + 255) / 256, 256, 0, stream>>>(Whh1, whh1b, GG * HH);

    // Layer 0: fp32 x input, store full bf16 h1 sequence.
    lstm_scan_kernel<DD, false, true><<<BB / 16, 1024, 0, stream>>>(
        x, wih0b, whh0b, bih0, bhh0, h1, nullptr);
    // Layer 1: bf16 h1 input, store only final fp32 hidden state.
    lstm_scan_kernel<HH, true, false><<<BB / 16, 1024, 0, stream>>>(
        h1, wih1b, whh1b, bih1, bhh1, nullptr, h2l);

    head_kernel<<<(BB + 63) / 64, 64, 0, stream>>>(h2l, Whead, bhead, (float*)d_out);
}
